// TransformerBlock_10204842295867
// MI455X (gfx1250) — matrix operand order repaired and checked by reading, not compiled, not run on hardware
//
#include <hip/hip_runtime.h>

// ============================================================================
// MI455X (gfx1250) transformer block, wave32 + WMMA bf16.
// Roofline: ~520 GFLOP vs ~300MB traffic @23.3TB/s -> compute bound; all
// GEMMs routed through v_wmma_f32_16x16x32_bf16 (f32 accum). Softmax attns
// use a flash-attention kernel (no TxT materialization); the softmax-free
// cross attention is reassociated as Q @ (K^T V) (exact, 8x fewer FLOPs).
// v3: GEMM uses 128x64 block tile, 2x2 register blocking per wave (4
// independent WMMA chains) and LDS double-buffering fed by the gfx1250
// async copy unit (global_load_async_to_lds_b128 + s_wait_asynccnt).
// ============================================================================

typedef __attribute__((ext_vector_type(16))) __bf16 v16bf;
typedef __attribute__((ext_vector_type(8)))  __bf16 v8bf;
typedef __attribute__((ext_vector_type(8)))  float  v8f;

#define DEV __device__ __forceinline__

DEV v8f wmma_bf16(v16bf a, v16bf b, v8f c) {
  return __builtin_amdgcn_wmma_f32_16x16x32_bf16(false, a, false, b, (short)0, c,
                                                 false, false);
}

// A-fragment (16x32 bf16): lane L holds row m=L&15; lanes 0-15
// carry K=[0..15], lanes 16-31 carry K=[16..31] -> two
// contiguous 8-element (16B) chunks per lane.
DEV v16bf load_afrag(const __bf16* rowp, int hi) {
  const int c0 = hi ? 16 : 0;
  const int c1 = hi ? 24 : 8;
  v8bf lo = *(const v8bf*)(rowp + c0);
  v8bf hb = *(const v8bf*)(rowp + c1);
  v16bf a;
#pragma unroll
  for (int i = 0; i < 8; ++i) { a[i] = lo[i]; a[8 + i] = hb[i]; }
  return a;
}

// gfx1250 async copy: 16B global -> LDS, tracked by ASYNCcnt (ISA 15.18.3
// opcode 98). LDS address = low 32 bits of the generic pointer (LDS aperture
// maps addr[31:0] to the LDS offset, ISA 10.2).
DEV void async_g2l_b128(const __bf16* gsrc, __bf16* ldst) {
  unsigned lws = (unsigned)(unsigned long long)ldst;
  asm volatile("global_load_async_to_lds_b128 %0, %1, off"
               :: "v"(lws), "v"(gsrc) : "memory");
}
DEV void wait_async0() { asm volatile("s_wait_asynccnt 0" ::: "memory"); }

// ---------------------------------------------------------------------------
// fp32 -> bf16 conversion (weights / raw activations)
// ---------------------------------------------------------------------------
__global__ void k_cvt_bf16(const float* __restrict__ x, __bf16* __restrict__ y,
                           long n) {
  long i = (long)blockIdx.x * blockDim.x + threadIdx.x;
  long stride = (long)gridDim.x * blockDim.x;
  for (; i < n; i += stride) y[i] = (__bf16)x[i];
}

// ---------------------------------------------------------------------------
// Generic strided WMMA GEMM:  C(m,n) = alpha * sum_k A(m,k) B(k,n) [+bias][relu]
//   A(m,k) = A[m*sAm + k*sAk],  B(k,n) = B[k*sBk + n*sBn]  (+ batch offsets)
//   gridDim.z = Bo*Hh batches; z -> (bo, hh) offsets.
// Requires M multiple of 128, N multiple of 64, K multiple of 32; every
// operand must be contiguous (stride 1, 16B-aligned base) in either k or m/n
// -- true for all call sites here.
// Block = 256 threads = 8 waves; tile 128x64; each wave owns a 32x32 subtile
// (2x2 register blocking -> 4 independent WMMA chains). LDS is double
// buffered: k-contiguous operands stream in via the async copy unit while
// the current tile is being consumed.
// ---------------------------------------------------------------------------
__global__ __launch_bounds__(256) void k_gemm_bf16(
    const __bf16* __restrict__ A, long sAm, long sAk, long sAh, long sAb,
    const __bf16* __restrict__ Bm, long sBk, long sBn, long sBh, long sBb,
    float* __restrict__ Cf, __bf16* __restrict__ Cb, long sCm, long sCh,
    long sCbt, const float* __restrict__ bias, int relu, float alpha, int M,
    int N, int K, int Hh) {
  __shared__ __attribute__((aligned(32))) __bf16 As[2][128 * 32];  // [m][k]
  __shared__ __attribute__((aligned(32))) __bf16 Bs[2][64 * 32];   // [n][k]

  const int tid = threadIdx.x;
  const int lane = tid & 31;
  const int wid = tid >> 5;
  const int lh = lane & 15;
  const int hi = lane >> 4;
  const int wr = wid & 3;    // 4 row groups of 32
  const int wc = wid >> 2;   // 2 col groups of 32
  const int tm = blockIdx.y * 128;
  const int tn = blockIdx.x * 64;
  const int z = blockIdx.z;
  const int bo = z / Hh, hh = z % Hh;

  const __bf16* Ab = A + (long)bo * sAb + (long)hh * sAh;
  const __bf16* Bb = Bm + (long)bo * sBb + (long)hh * sBh;

  // stage one 128x32 A tile + 64x32 B tile into LDS buffer `buf`
  auto stage = [&](int buf, int k0) {
    if (sAk == 1) {                    // k-contiguous: async copy unit
#pragma unroll
      for (int t = 0; t < 2; ++t) {
        const int c = tid + t * 256;   // 512 x 16B chunks
        const int mm = c >> 2, kc = (c & 3) * 8;
        async_g2l_b128(Ab + (long)(tm + mm) * sAm + (k0 + kc),
                       &As[buf][mm * 32 + kc]);
      }
    } else {                           // m-contiguous: b128 + transpose store
#pragma unroll
      for (int t = 0; t < 2; ++t) {
        const int c = tid + t * 256;
        const int cm = (c & 15) * 8, kk = c >> 4;
        v8bf av = *(const v8bf*)(Ab + (long)(tm + cm) + (long)(k0 + kk) * sAk);
#pragma unroll
        for (int j = 0; j < 8; ++j) As[buf][(cm + j) * 32 + kk] = av[j];
      }
    }
    if (sBk == 1) {                    // k-contiguous (NT weights): async
      const int nn = tid >> 2, kc = (tid & 3) * 8;
      async_g2l_b128(Bb + (long)(tn + nn) * sBn + (k0 + kc),
                     &Bs[buf][nn * 32 + kc]);
    } else {                           // n-contiguous: b128 + transpose store
      const int cn = (tid & 7) * 8, kk = tid >> 3;
      v8bf bv = *(const v8bf*)(Bb + (long)(k0 + kk) * sBk + (tn + cn));
#pragma unroll
      for (int j = 0; j < 8; ++j) Bs[buf][(cn + j) * 32 + kk] = bv[j];
    }
  };

  v8f acc[2][2];
#pragma unroll
  for (int sr = 0; sr < 2; ++sr)
#pragma unroll
    for (int sc = 0; sc < 2; ++sc)
#pragma unroll
      for (int i = 0; i < 8; ++i) acc[sr][sc][i] = 0.f;

  stage(0, 0);
  for (int k0 = 0; k0 < K; k0 += 32) {
    const int cur = (k0 >> 5) & 1;
    wait_async0();       // drain this wave's async copies into `cur`
    __syncthreads();     // tile `cur` visible to all; `cur^1` free to refill
    if (k0 + 32 < K) stage(cur ^ 1, k0 + 32);

    v16bf af0 = load_afrag(&As[cur][(wr * 32 + lh) * 32], hi);
    v16bf af1 = load_afrag(&As[cur][(wr * 32 + 16 + lh) * 32], hi);
    v16bf bf0 = *(const v16bf*)(&Bs[cur][(wc * 32 + lh) * 32 + hi * 16]);
    v16bf bf1 = *(const v16bf*)(&Bs[cur][(wc * 32 + 16 + lh) * 32 + hi * 16]);
    acc[0][0] = wmma_bf16(af0, bf0, acc[0][0]);
    acc[0][1] = wmma_bf16(af0, bf1, acc[0][1]);
    acc[1][0] = wmma_bf16(af1, bf0, acc[1][0]);
    acc[1][1] = wmma_bf16(af1, bf1, acc[1][1]);
  }

  const long cbase = (long)bo * sCbt + (long)hh * sCh;
#pragma unroll
  for (int sr = 0; sr < 2; ++sr)
#pragma unroll
    for (int sc = 0; sc < 2; ++sc) {
      v8f a = acc[sr][sc];
#pragma unroll
      for (int r = 0; r < 8; ++r) {
        const int row = tm + wr * 32 + sr * 16 + r + hi * 8;  // C/D 7.12.2
        const int col = tn + wc * 32 + sc * 16 + lh;
        float v = a[r] * alpha + (bias ? bias[col] : 0.f);
        if (relu) v = fmaxf(v, 0.f);
        const long off = cbase + (long)row * sCm + col;
        if (Cf) Cf[off] = v;
        if (Cb) Cb[off] = (__bf16)v;
      }
    }
}

// ---------------------------------------------------------------------------
// Flash attention (softmax), causal or full. Layout: q/k/v bf16 [B,T,H*DH]
// Block = 128 threads (4 waves); each wave owns 16 query rows; block covers
// 64 rows; key loop in steps of 32. K tile is copied with async global->LDS
// (ASYNCcnt); V tile is vector-loaded and transposed into [d][s] so the P.V
// B-fragments are contiguous in k=s.
// ---------------------------------------------------------------------------
__global__ __launch_bounds__(128) void k_flash(
    const __bf16* __restrict__ Qb, const __bf16* __restrict__ Kb,
    const __bf16* __restrict__ Vb, float* __restrict__ O, int T_, int Dm,
    int DHh, int causal, float scale) {
  __shared__ __attribute__((aligned(32))) __bf16 Ks[32 * 256];   // [s][e]
  __shared__ __attribute__((aligned(32))) __bf16 Vs[256 * 32];   // [d][s]
  __shared__ __attribute__((aligned(32))) __bf16 Ps[4 * 16 * 32];

  const int tid = threadIdx.x;
  const int wid = tid >> 5;
  const int lane = tid & 31;
  const int lh = lane & 15;
  const int hi = lane >> 4;
  const int b = blockIdx.z, h = blockIdx.y;
  const int qblk = blockIdx.x * 64;
  const int q0 = qblk + wid * 16;
  const long base = (long)b * T_ * Dm + (long)h * DHh;

  // Q fragments for this wave's 16 rows (DH=256 -> 8 K-chunks of 32)
  v16bf qf[8];
  {
    const __bf16* qrow = Qb + base + (long)(q0 + lh) * Dm;
#pragma unroll
    for (int kk = 0; kk < 8; ++kk) qf[kk] = load_afrag(qrow + kk * 32, hi);
  }

  v8f acc[16];
#pragma unroll
  for (int dt = 0; dt < 16; ++dt)
#pragma unroll
    for (int r = 0; r < 8; ++r) acc[dt][r] = 0.f;

  float mrow[8], lrow[8];
#pragma unroll
  for (int r = 0; r < 8; ++r) { mrow[r] = -3e38f; lrow[r] = 0.f; }

  int kend = causal ? (qblk + 64) : T_;
  if (kend > T_) kend = T_;

  for (int s0 = 0; s0 < kend; s0 += 32) {
    // cooperative K/V tile load: 1024 x 16B chunks, 8 per thread
#pragma unroll
    for (int c = tid; c < 1024; c += 128) {
      const int s = c >> 5, e0 = (c & 31) * 8;
      const long g = base + (long)(s0 + s) * Dm + e0;
      async_g2l_b128(Kb + g, &Ks[s * 256 + e0]);     // K: async copy to LDS
      v8bf vv = *(const v8bf*)(Vb + g);              // V: load + transpose
#pragma unroll
      for (int j = 0; j < 8; ++j) Vs[(e0 + j) * 32 + s] = vv[j];
    }
    wait_async0();           // each wave drains its own async copies
    __syncthreads();

    // S = scale * Q K^T : two 16x16 tiles covering 32 keys
    v8f S[2];
#pragma unroll
    for (int half = 0; half < 2; ++half) {
      v8f s8;
#pragma unroll
      for (int r = 0; r < 8; ++r) s8[r] = 0.f;
#pragma unroll
      for (int kk = 0; kk < 8; ++kk) {
        v16bf bf =
            *(const v16bf*)(&Ks[(half * 16 + lh) * 256 + kk * 32 + hi * 16]);
        s8 = wmma_bf16(qf[kk], bf, s8);
      }
#pragma unroll
      for (int r = 0; r < 8; ++r) {
        float v = s8[r] * scale;
        if (causal) {
          const int qi = q0 + r + hi * 8;
          const int kj = s0 + half * 16 + lh;
          if (kj > qi) v = -1e30f;
        }
        s8[r] = v;
      }
      S[half] = s8;
    }

    // online softmax: row max / row sum across 16 lanes of each half-group
    float tmax[8];
#pragma unroll
    for (int r = 0; r < 8; ++r) tmax[r] = fmaxf(S[0][r], S[1][r]);
#pragma unroll
    for (int m = 1; m < 16; m <<= 1)
#pragma unroll
      for (int r = 0; r < 8; ++r)
        tmax[r] = fmaxf(tmax[r], __shfl_xor(tmax[r], m, 32));

    float corr[8], rsum[8];
#pragma unroll
    for (int r = 0; r < 8; ++r) {
      const float nm = fmaxf(mrow[r], tmax[r]);
      corr[r] = __expf(mrow[r] - nm);
      mrow[r] = nm;
      const float p0 = __expf(S[0][r] - nm);
      const float p1 = __expf(S[1][r] - nm);
      S[0][r] = p0; S[1][r] = p1;
      rsum[r] = p0 + p1;
    }
#pragma unroll
    for (int m = 1; m < 16; m <<= 1)
#pragma unroll
      for (int r = 0; r < 8; ++r) rsum[r] += __shfl_xor(rsum[r], m, 32);
#pragma unroll
    for (int r = 0; r < 8; ++r) lrow[r] = lrow[r] * corr[r] + rsum[r];
#pragma unroll
    for (int dt = 0; dt < 16; ++dt)
#pragma unroll
      for (int r = 0; r < 8; ++r) acc[dt][r] *= corr[r];

    // re-layout P (lane=col,vgpr=row) -> A fragment (lane=row) via wave LDS
    __bf16* pw = &Ps[wid * 16 * 32];
#pragma unroll
    for (int half = 0; half < 2; ++half)
#pragma unroll
      for (int r = 0; r < 8; ++r)
        pw[(r + hi * 8) * 32 + half * 16 + lh] = (__bf16)S[half][r];
    asm volatile("s_wait_dscnt 0" ::: "memory");   // wave-local LDS RAW

    v16bf pf = load_afrag(pw + lh * 32, hi);

    // acc += P @ V  (16 output d-tiles of 16)
#pragma unroll
    for (int dt = 0; dt < 16; ++dt) {
      v16bf bv = *(const v16bf*)(&Vs[(dt * 16 + lh) * 32 + hi * 16]);
      acc[dt] = wmma_bf16(pf, bv, acc[dt]);
    }
    __syncthreads();
  }

  // O = acc / l, fp32 out (consumed by residual+LN)
#pragma unroll
  for (int dt = 0; dt < 16; ++dt)
#pragma unroll
    for (int r = 0; r < 8; ++r) {
      const int row = q0 + r + hi * 8;
      const int col = h * DHh + dt * 16 + lh;
      O[((long)b * T_ + row) * Dm + col] = acc[dt][r] / lrow[r];
    }
}

// ---------------------------------------------------------------------------
// y = LayerNorm(x + r) * g + b ; writes fp32 (and optional bf16 copy for the
// next WMMA GEMM). One block per row (D=1024).
// ---------------------------------------------------------------------------
__global__ __launch_bounds__(256) void k_add_ln(
    const float* __restrict__ x, const float* __restrict__ r,
    const float* __restrict__ g, const float* __restrict__ be,
    float* __restrict__ yf, __bf16* __restrict__ yb, int Dm) {
  __shared__ float buf[1024];
  __shared__ float red[256];
  const int row = blockIdx.x, tid = threadIdx.x;
  const long ro = (long)row * Dm;

  float s = 0.f;
  for (int i = tid; i < Dm; i += 256) {
    const float v = x[ro + i] + r[ro + i];
    buf[i] = v;
    s += v;
  }
  red[tid] = s;
  __syncthreads();
  for (int st = 128; st > 0; st >>= 1) {
    if (tid < st) red[tid] += red[tid + st];
    __syncthreads();
  }
  const float mean = red[0] / Dm;
  __syncthreads();

  float vs = 0.f;
  for (int i = tid; i < Dm; i += 256) {
    const float d = buf[i] - mean;
    vs += d * d;
  }
  red[tid] = vs;
  __syncthreads();
  for (int st = 128; st > 0; st >>= 1) {
    if (tid < st) red[tid] += red[tid + st];
    __syncthreads();
  }
  const float inv = rsqrtf(red[0] / Dm + 1e-5f);

  for (int i = tid; i < Dm; i += 256) {
    const float v = (buf[i] - mean) * inv * g[i] + be[i];
    yf[ro + i] = v;
    if (yb) yb[ro + i] = (__bf16)v;
  }
}

// ===========================================================================
// Host orchestration
// ===========================================================================
extern "C" void kernel_launch(void* const* d_in, const int* in_sizes, int n_in,
                              void* d_out, int out_size, void* d_ws,
                              size_t ws_size, hipStream_t stream) {
  (void)in_sizes; (void)n_in; (void)out_size; (void)ws_size;
  constexpr long Bc = 4, Tc = 2048, Dc = 1024, Hc = 4, DHc = 256;
  constexpr long BT = Bc * Tc;          // 8192 rows
  constexpr long BTD = BT * Dc;         // 8,388,608
  constexpr long FFND = 4 * Dc;         // 4096
  const float scale = 0.0625f;          // DH^-0.5

  const float* context = (const float*)d_in[0];
  const float* X       = (const float*)d_in[1];
  const float* dec_wq  = (const float*)d_in[2];
  const float* dec_wk  = (const float*)d_in[3];
  const float* dec_wv  = (const float*)d_in[4];
  const float* x_wq    = (const float*)d_in[5];
  const float* x_wk    = (const float*)d_in[6];
  const float* x_wv    = (const float*)d_in[7];
  const float* enc_wq  = (const float*)d_in[8];
  const float* enc_wk  = (const float*)d_in[9];
  const float* enc_wv  = (const float*)d_in[10];
  const float* dec_w1  = (const float*)d_in[11];
  const float* dec_b1  = (const float*)d_in[12];
  const float* dec_w2  = (const float*)d_in[13];
  const float* dec_b2  = (const float*)d_in[14];
  const float* enc_w1  = (const float*)d_in[15];
  const float* enc_b1  = (const float*)d_in[16];
  const float* enc_w2  = (const float*)d_in[17];
  const float* enc_b2  = (const float*)d_in[18];
  const float* d_ln1g = (const float*)d_in[19]; const float* d_ln1b = (const float*)d_in[20];
  const float* d_ln2g = (const float*)d_in[21]; const float* d_ln2b = (const float*)d_in[22];
  const float* d_ln3g = (const float*)d_in[23]; const float* d_ln3b = (const float*)d_in[24];
  const float* e_ln1g = (const float*)d_in[25]; const float* e_ln1b = (const float*)d_in[26];
  const float* e_ln2g = (const float*)d_in[27]; const float* e_ln2b = (const float*)d_in[28];

  float* out_c = (float*)d_out;          // c  : first B*T*D
  float* out_x = (float*)d_out + BTD;    // out: second B*T*D

  // ---- bump allocator over d_ws (total ~313 MB) ----
  char* p = (char*)d_ws;
  auto alloc = [&](size_t bytes) -> void* {
    void* r = (void*)p;
    p += (bytes + 255) & ~(size_t)255;
    return r;
  };
  float* F0 = (float*)alloc(BTD * 4);                 // attn/ffn temp
  float* F1 = (float*)alloc(BTD * 4);                 // x1 fp32
  float* F2 = (float*)alloc(BTD * 4);                 // c_pre fp32 -> x2 fp32
  float* KTV = (float*)alloc(Bc * Hc * DHc * DHc * 4);
  __bf16* HA = (__bf16*)alloc(BTD * 2);               // xb -> cb -> x2b
  __bf16* HB = (__bf16*)alloc(BTD * 2);               // ctxb -> cb2
  __bf16* HC = (__bf16*)alloc(BTD * 2);               // x1b
  __bf16* HQ = (__bf16*)alloc(BTD * 2);
  __bf16* HK = (__bf16*)alloc(BTD * 2);
  __bf16* HV = (__bf16*)alloc(BTD * 2);
  __bf16* HID = (__bf16*)alloc(BT * FFND * 2);        // FFN hidden (bf16, post-relu)
  __bf16* KTVB = (__bf16*)alloc(Bc * Hc * DHc * DHc * 2);
  __bf16* WQD = (__bf16*)alloc(Hc * DHc * DHc * 2);
  __bf16* WKD = (__bf16*)alloc(Hc * DHc * DHc * 2);
  __bf16* WVD = (__bf16*)alloc(Hc * DHc * DHc * 2);
  __bf16* WQX = (__bf16*)alloc(Hc * DHc * DHc * 2);
  __bf16* WKX = (__bf16*)alloc(Hc * DHc * DHc * 2);
  __bf16* WVX = (__bf16*)alloc(Hc * DHc * DHc * 2);
  __bf16* WQE = (__bf16*)alloc(Hc * DHc * DHc * 2);
  __bf16* WKE = (__bf16*)alloc(Hc * DHc * DHc * 2);
  __bf16* WVE = (__bf16*)alloc(Hc * DHc * DHc * 2);
  __bf16* W1D = (__bf16*)alloc(FFND * Dc * 2);
  __bf16* W2D = (__bf16*)alloc(FFND * Dc * 2);
  __bf16* W1E = (__bf16*)alloc(FFND * Dc * 2);
  __bf16* W2E = (__bf16*)alloc(FFND * Dc * 2);

  auto cvt = [&](const float* src, __bf16* dst, long n) {
    long nb = (n + 255) / 256;
    if (nb > 4096) nb = 4096;
    k_cvt_bf16<<<(int)nb, 256, 0, stream>>>(src, dst, n);
  };
  auto gemm = [&](const __bf16* A, long sAm, long sAk, long sAh, long sAb,
                  const __bf16* Bm, long sBk, long sBn, long sBh, long sBb,
                  float* Cf, __bf16* Cb, long sCm, long sCh, long sCbt,
                  const float* bias, int relu, float alpha, int M, int N,
                  int K, int Hh, int Z) {
    dim3 g(N / 64, M / 128, Z);
    k_gemm_bf16<<<g, 256, 0, stream>>>(A, sAm, sAk, sAh, sAb, Bm, sBk, sBn,
                                       sBh, sBb, Cf, Cb, sCm, sCh, sCbt, bias,
                                       relu, alpha, M, N, K, Hh);
  };
  // heads projection: out[b,t,h,:] = x[b,t,h,:] @ W[h]^T  (batched over H)
  auto proj = [&](const __bf16* xb, const __bf16* W, __bf16* y) {
    gemm(xb, Dc, 1, DHc, 0, W, 1, DHc, DHc * DHc, 0, nullptr, y, Dc, DHc, 0,
         nullptr, 0, 1.f, (int)BT, (int)DHc, (int)DHc, (int)Hc, (int)Hc);
  };
  auto flash = [&](const __bf16* q, const __bf16* k, const __bf16* v,
                   float* o, int causal) {
    dim3 g((int)(Tc / 64), (int)Hc, (int)Bc);
    k_flash<<<g, 128, 0, stream>>>(q, k, v, o, (int)Tc, (int)Dc, (int)DHc,
                                   causal, scale);
  };
  auto addln = [&](const float* x, const float* r, const float* g_,
                   const float* b_, float* yf, __bf16* yb) {
    k_add_ln<<<(int)BT, 256, 0, stream>>>(x, r, g_, b_, yf, yb, (int)Dc);
  };

  // ---- weight / input conversion to bf16 ----
  cvt(dec_wq, WQD, Hc * DHc * DHc); cvt(dec_wk, WKD, Hc * DHc * DHc);
  cvt(dec_wv, WVD, Hc * DHc * DHc); cvt(x_wq, WQX, Hc * DHc * DHc);
  cvt(x_wk, WKX, Hc * DHc * DHc);   cvt(x_wv, WVX, Hc * DHc * DHc);
  cvt(enc_wq, WQE, Hc * DHc * DHc); cvt(enc_wk, WKE, Hc * DHc * DHc);
  cvt(enc_wv, WVE, Hc * DHc * DHc);
  cvt(dec_w1, W1D, FFND * Dc); cvt(dec_w2, W2D, FFND * Dc);
  cvt(enc_w1, W1E, FFND * Dc); cvt(enc_w2, W2E, FFND * Dc);
  cvt(X, HA, BTD);        // xb
  cvt(context, HB, BTD);  // ctxb

  // ---- decoder masked self-attention + LN1 ----
  proj(HA, WQD, HQ); proj(HA, WKD, HK); proj(HA, WVD, HV);
  flash(HQ, HK, HV, F0, /*causal=*/1);
  addln(X, F0, d_ln1g, d_ln1b, F1, HC);          // x1 (fp32 F1, bf16 HC)

  // ---- encoder: self-attention + LN1, FFN + LN2 -> c ----
  proj(HB, WQE, HQ); proj(HB, WKE, HK); proj(HB, WVE, HV);
  flash(HQ, HK, HV, F0, /*causal=*/0);
  addln(context, F0, e_ln1g, e_ln1b, F2, HA);    // c_pre (fp32 F2, bf16 HA)
  gemm(HA, Dc, 1, 0, 0, W1E, 1, Dc, 0, 0, nullptr, HID, FFND, 0, 0, enc_b1, 1,
       1.f, (int)BT, (int)FFND, (int)Dc, 1, 1);
  gemm(HID, FFND, 1, 0, 0, W2E, 1, FFND, 0, 0, F0, nullptr, Dc, 0, 0, enc_b2,
       0, 1.f, (int)BT, (int)Dc, (int)FFND, 1, 1);
  addln(F2, F0, e_ln2g, e_ln2b, out_c, HB);      // c -> d_out[0:BTD], bf16 HB

  // ---- cross attention (no softmax): o = scale * Q (K^T V) ----
  proj(HC, WQX, HQ);   // q from x1
  proj(HB, WKX, HK);   // k from c
  proj(HB, WVX, HV);   // v from c
  // KTV[b,h] = scale * K^T V : M=DH(e), N=DH(d), K=T(s), batched over B*H
  gemm(HK, 1, Dc, DHc, Tc * Dc, HV, Dc, 1, DHc, Tc * Dc, KTV, KTVB, DHc,
       DHc * DHc, Hc * DHc * DHc, nullptr, 0, scale, (int)DHc, (int)DHc,
       (int)Tc, (int)Hc, (int)(Bc * Hc));
  // o[b,:,h,:] = Q[b,:,h,:] @ KTV[b,h]
  gemm(HQ, Dc, 1, DHc, Tc * Dc, KTVB, DHc, 1, DHc * DHc, Hc * DHc * DHc, F0,
       nullptr, Dc, DHc, Tc * Dc, nullptr, 0, 1.f, (int)Tc, (int)DHc,
       (int)DHc, (int)Hc, (int)(Bc * Hc));
  addln(F1, F0, d_ln2g, d_ln2b, F2, HA);         // x2 (fp32 F2, bf16 HA)

  // ---- decoder FFN + LN3 -> out ----
  gemm(HA, Dc, 1, 0, 0, W1D, 1, Dc, 0, 0, nullptr, HID, FFND, 0, 0, dec_b1, 1,
       1.f, (int)BT, (int)FFND, (int)Dc, 1, 1);
  gemm(HID, FFND, 1, 0, 0, W2D, 1, FFND, 0, 0, F0, nullptr, Dc, 0, 0, dec_b2,
       0, 1.f, (int)BT, (int)Dc, (int)FFND, 1, 1);
  addln(F2, F0, d_ln3g, d_ln3b, out_x, nullptr); // out -> d_out[BTD:2*BTD]
}
